// BertSelfAttention_11227044512079
// MI455X (gfx1250) — matrix operand order repaired and checked by reading, not compiled, not run on hardware
//
#include <hip/hip_runtime.h>
#include <hip/hip_bf16.h>
#include <math.h>

// ---------------------------------------------------------------------------
// BERT self-attention with relative_key_query bias, CDNA5 (gfx1250), wave32.
// bf16 V_WMMA_F32_16X16X32_BF16 everywhere, fp32 accumulation.
// Bulk LDS staging uses GLOBAL_LOAD_ASYNC_TO_LDS_B128 (ASYNCcnt path).
// ---------------------------------------------------------------------------

typedef __attribute__((ext_vector_type(8)))  float   v8f;
typedef __attribute__((ext_vector_type(16))) __bf16  v16bf;

#define NB   16      // batch
#define NS   512     // seq len
#define ND   1024    // hidden
#define NH   20      // total heads (16 base + 4 domain)
#define HD   64      // head dim
#define NT   1280    // NH*HD
#define MAXP 512

union U32 { unsigned int u; __bf16 h[2]; };

static __device__ __forceinline__ v8f zero8() {
  v8f z = {0.f, 0.f, 0.f, 0.f, 0.f, 0.f, 0.f, 0.f};
  return z;
}

// D = A(16x32 bf16) * B(32x16 bf16) + C, fp32 accum
static __device__ __forceinline__ v8f wmma_bf16(v16bf a, v16bf b, v8f c) {
  return __builtin_amdgcn_wmma_f32_16x16x32_bf16(
      /*neg_a=*/false, a, /*neg_b=*/false, b,
      /*c_mod=*/(short)0, c, /*reuse_a=*/false, /*reuse_b=*/false);
}

// Async copy of 16 bytes per lane: global -> LDS, tracked by ASYNCcnt.
// LDS address = low 32 bits of the generic pointer (ISA 10.2: LDS aperture
// truncates to addr[31:0]); global address in a VGPR pair, SADDR = off.
static __device__ __forceinline__ void async_b128(const void* lds_ptr, const void* gptr) {
  unsigned lds_off = (unsigned)(size_t)lds_ptr;
  unsigned long long ga = (unsigned long long)(size_t)gptr;
  asm volatile("global_load_async_to_lds_b128 %0, %1, off"
               :: "v"(lds_off), "v"(ga) : "memory");
}
static __device__ __forceinline__ void async_wait0() {
  asm volatile("s_wait_asynccnt 0" ::: "memory");
}

// A-matrix 16x32 bf16 fragment (ISA 7.12.2):
//   lanes 0-15 : row=lane,    elems 0-7 -> K0-7,   elems 8-15 -> K16-23
//   lanes16-31 : row=lane-16, elems 0-7 -> K8-15,  elems 8-15 -> K24-31
static __device__ __forceinline__ v16bf frag_a(const __bf16* base, int stride, int lane) {
  const int row = lane & 15, hf = lane >> 4;
  const __bf16* p = base + row * stride + 8 * hf;
  v16bf f;
  *(uint4*)(&f)       = *(const uint4*)(p);        // K base .. base+7
  *((uint4*)(&f) + 1) = *(const uint4*)(p + 16);   // K base+16 .. base+23
  return f;
}

// B-matrix 32x16 bf16 fragment: lane n<16 holds col n, K0-7,K16-23; lanes 16-31 K8-15,K24-31.
// Source layout is K-major per column: elem(k,col) at base + col*stride + k.
static __device__ __forceinline__ v16bf frag_b(const __bf16* base, int stride, int lane) {
  const int col = lane & 15, kb = 8 * (lane >> 4);
  const __bf16* p = base + col * stride + kb;
  v16bf f;
  *(uint4*)(&f)       = *(const uint4*)(p);
  *((uint4*)(&f) + 1) = *(const uint4*)(p + 16);
  return f;
}

// ---------------------------------------------------------------------------
// Kernel 0: pack fp32 inputs to bf16 (hs, concatenated weights, dist_emb)
// ---------------------------------------------------------------------------
__global__ void pack_x_kernel(const float* __restrict__ x, __bf16* __restrict__ o, int n) {
  int i = blockIdx.x * 256 + threadIdx.x;
  if (i < n) o[i] = (__bf16)x[i];
}

__global__ void pack_w_kernel(const float* __restrict__ Wq, const float* __restrict__ Wk,
                              const float* __restrict__ Wv, const float* __restrict__ Wqd,
                              const float* __restrict__ Wkd, const float* __restrict__ Wvd,
                              __bf16* __restrict__ o) {
  int i = blockIdx.x * 256 + threadIdx.x;
  if (i >= 3 * ND * NT) return;
  int g  = i / (ND * NT);
  int r  = i - g * (ND * NT);
  int kk = r / NT;
  int n  = r - kk * NT;
  const float* Wb = (g == 0) ? Wq  : ((g == 1) ? Wk  : Wv);
  const float* Wd = (g == 0) ? Wqd : ((g == 1) ? Wkd : Wvd);
  float v = (n < ND) ? Wb[kk * ND + n] : Wd[kk * 256 + (n - ND)];
  o[i] = (__bf16)v;
}

__global__ void pack_b_kernel(const float* __restrict__ bq, const float* __restrict__ bk,
                              const float* __restrict__ bv, const float* __restrict__ bqd,
                              const float* __restrict__ bkd, const float* __restrict__ bvd,
                              float* __restrict__ o) {
  int i = blockIdx.x * 256 + threadIdx.x;
  if (i >= 3 * NT) return;
  int g = i / NT, n = i - g * NT;
  const float* Bb = (g == 0) ? bq  : ((g == 1) ? bk  : bv);
  const float* Bd = (g == 0) ? bqd : ((g == 1) ? bkd : bvd);
  o[i] = (n < ND) ? Bb[n] : Bd[n - ND];
}

// dist_emb [1023,64] -> bf16, padded to 1024 rows (band GEMM over-reads row 1023)
__global__ void pack_e_kernel(const float* __restrict__ e, __bf16* __restrict__ o) {
  int i = blockIdx.x * 256 + threadIdx.x;
  if (i < 1024 * HD) o[i] = (i < (2 * MAXP - 1) * HD) ? (__bf16)e[i] : (__bf16)0.f;
}

// ---------------------------------------------------------------------------
// Kernel 1: fused QKV projection.  [8192 x 1024] @ [1024 x 1280] for g=0,1,2.
// 128x128 tile / workgroup, 8 waves in 2x4 grid, K-step 32, bf16 WMMA.
// A tile staged via async-to-LDS; B tile staged with a bf16 transpose.
// Output written head-major: {Q,K,V}bf[b][h][s][d] bf16.
// ---------------------------------------------------------------------------
__global__ __launch_bounds__(256) void qkv_gemm_kernel(
    const __bf16* __restrict__ Xbf, const __bf16* __restrict__ Wcat,
    const float* __restrict__ Bcat,
    __bf16* __restrict__ Qbf, __bf16* __restrict__ Kbf, __bf16* __restrict__ Vbf) {
  __shared__ __align__(16) __bf16 sA[128 * 32];   // [row][k] row-major
  __shared__ __align__(16) __bf16 sBt[128 * 32];  // [n][k]  K-major per column

  const int tid = threadIdx.x, lane = tid & 31, wv = tid >> 5;
  const int wm = wv & 1, wn = wv >> 1;            // 2 x 4 wave grid
  const int hf = lane >> 4, ln = lane & 15;
  const int m0 = blockIdx.x * 128, n0 = blockIdx.y * 128, g = blockIdx.z;

  const char* Xbytes = (const char*)Xbf;                 // 2048 B / row
  const unsigned int* Wdw = (const unsigned int*)Wcat;   // 640 dw / row

  v8f acc[4][2];
#pragma unroll
  for (int ti = 0; ti < 4; ++ti)
#pragma unroll
    for (int tj = 0; tj < 2; ++tj) acc[ti][tj] = zero8();

  for (int kb = 0; kb < ND; kb += 32) {
    __syncthreads();
    // stage A tile 128x32 bf16 via async-to-LDS (512 x 16B chunks, 4 per row)
#pragma unroll
    for (int i = 0; i < 2; ++i) {
      int c = tid + i * 256;
      int row = c >> 2, c16 = c & 3;
      async_b128((const char*)sA + c * 16,
                 Xbytes + (size_t)(m0 + row) * 2048 + (size_t)kb * 2 + c16 * 16);
    }
    // stage B tile 32x128 transposed to [n][k] (2048 dwords -> 4096 b16 stores)
#pragma unroll
    for (int i = 0; i < 8; ++i) {
      int idx = tid + i * 256;
      int k = idx >> 6, n2 = idx & 63;
      U32 c;
      c.u = Wdw[(size_t)(g * ND + kb + k) * 640 + (n0 >> 1) + n2];
      sBt[(2 * n2) * 32 + k]     = c.h[0];
      sBt[(2 * n2 + 1) * 32 + k] = c.h[1];
    }
    if (kb + 32 < ND)  // speculative prefetch of next A tile -> global_prefetch_b8
      __builtin_prefetch(Xbytes + (size_t)(m0 + (tid >> 1)) * 2048 + (size_t)(kb + 32) * 2, 0, 0);
    async_wait0();
    __syncthreads();

    v16bf bf2[2];
#pragma unroll
    for (int tj = 0; tj < 2; ++tj)
      bf2[tj] = frag_b(sBt + (32 * wn + 16 * tj) * 32, 32, lane);
#pragma unroll
    for (int ti = 0; ti < 4; ++ti) {
      v16bf af = frag_a(sA + (64 * wm + 16 * ti) * 32, 32, lane);
#pragma unroll
      for (int tj = 0; tj < 2; ++tj) acc[ti][tj] = wmma_bf16(af, bf2[tj], acc[ti][tj]);
    }
  }

  __bf16* dst = (g == 0) ? Qbf : ((g == 1) ? Kbf : Vbf);
#pragma unroll
  for (int ti = 0; ti < 4; ++ti)
#pragma unroll
    for (int tj = 0; tj < 2; ++tj)
#pragma unroll
      for (int i = 0; i < 8; ++i) {
        int m = m0 + 64 * wm + 16 * ti + i + 8 * hf;
        int n = n0 + 32 * wn + 16 * tj + ln;
        float v = acc[ti][tj][i] + Bcat[g * NT + n];
        int bb = m >> 9, ss = m & 511, hh = n >> 6, dd = n & 63;
        dst[(((size_t)bb * NH + hh) * NS + ss) * HD + dd] = (__bf16)v;
      }
}

// ---------------------------------------------------------------------------
// Kernel 2: attention with banded relative-position bias, flash softmax.
// Workgroup = (b, h, 128-row l-block). 8 waves; wave w owns rows [16w,16w+16).
// Per r-tile of 64: async-stage K, transpose-stage V^T in LDS, S = Q K^T
// (WMMA), band GEMMs TQ = Q_block @ E_band^T and TK = K_tile @ E_band^T
// (WMMA, band-limited tile loops), bias gather via dd = a - b + 63,
// online softmax in-wave, ctx += P @ V (WMMA, P staged via per-wave LDS).
// ---------------------------------------------------------------------------
#define BAND 192  // 128 + 64 - 1 rounded to 16
#define ATTN_SMEM_BYTES ((128*64 + 64*64 + 64*64 + 128*BAND + 64*BAND + 8*16*64) * 2)

__global__ __launch_bounds__(256) void attn_kernel(
    const __bf16* __restrict__ Qbf, const __bf16* __restrict__ Kbf,
    const __bf16* __restrict__ Vbf, const __bf16* __restrict__ Ebf,
    const float* __restrict__ amask, float* __restrict__ out) {
  extern __shared__ __align__(16) char smem[];
  __bf16* sQ  = (__bf16*)smem;          // [128][64]
  __bf16* sK  = sQ  + 128 * 64;         // [64][64]
  __bf16* sVt = sK  + 64 * 64;          // [64 d][64 r]  (V transposed)
  __bf16* sTQ = sVt + 64 * 64;          // [128][BAND]
  __bf16* sTK = sTQ + 128 * BAND;       // [64][BAND]
  __bf16* sP  = sTK + 64 * BAND;        // [8 waves][16][64]

  const int l0 = blockIdx.x * 128, h = blockIdx.y, b = blockIdx.z;
  const int tid = threadIdx.x, w = tid >> 5, lane = tid & 31;
  const int hf = lane >> 4, ln = lane & 15;

  const __bf16* Qp = Qbf + ((size_t)(b * NH + h) * NS + l0) * HD;
  const __bf16* Kp = Kbf + ((size_t)(b * NH + h) * NS) * HD;
  const __bf16* Vp = Vbf + ((size_t)(b * NH + h) * NS) * HD;

  // stage Q block (128x64 bf16 = 16KB) via async-to-LDS
#pragma unroll
  for (int i = 0; i < 4; ++i) {
    int c = tid + i * 256;
    async_b128((const char*)sQ + c * 16, (const char*)Qp + c * 16);
  }
  async_wait0();
  __syncthreads();

  // this wave's Q A-fragments (reused for S and for the TQ band GEMM)
  v16bf qa0 = frag_a(sQ + (16 * w) * HD + 0,  HD, lane);
  v16bf qa1 = frag_a(sQ + (16 * w) * HD + 32, HD, lane);

  float m_run[8], l_run[8];
  v8f ctx[4];
#pragma unroll
  for (int i = 0; i < 8; ++i) { m_run[i] = -__builtin_inff(); l_run[i] = 0.f; }
#pragma unroll
  for (int ct = 0; ct < 4; ++ct) ctx[ct] = zero8();

  for (int rt = 0; rt < 8; ++rt) {
    const int r0 = rt * 64;
    const int dmin = l0 - r0 + 448;  // band base distance, always in [0, 832]
    __syncthreads();  // protect sK/sVt/sTK from previous iteration's readers

    // stage K tile (row-major, async) and V tile transposed (manual)
#pragma unroll
    for (int i = 0; i < 2; ++i) {
      int c = tid + i * 256;
      async_b128((const char*)sK + c * 16, (const char*)Kp + (size_t)r0 * 128 + c * 16);
    }
    {
      const unsigned int* vs = (const unsigned int*)(Vp + (size_t)r0 * HD);
#pragma unroll
      for (int i = 0; i < 8; ++i) {
        int idx = tid + i * 256;
        U32 c; c.u = vs[idx];
        int r = idx >> 5, d2 = (idx & 31) * 2;
        sVt[d2 * 64 + r]       = c.h[0];
        sVt[(d2 + 1) * 64 + r] = c.h[1];
      }
    }
    async_wait0();
    __syncthreads();

    // ---- S = Q K^T (this wave's 16 rows x 64 cols) ----
    v8f sfr[4];
#pragma unroll
    for (int ct = 0; ct < 4; ++ct) {
      v8f a = zero8();
      a = wmma_bf16(qa0, frag_b(sK + (16 * ct) * HD + 0,  HD, lane), a);
      a = wmma_bf16(qa1, frag_b(sK + (16 * ct) * HD + 32, HD, lane), a);
      sfr[ct] = a;
    }

    // ---- TQ band: rows 16w..16w+15 need dd in [16w, 16w+78] -> jt = w..w+4 ----
#pragma unroll
    for (int jj = 0; jj < 5; ++jj) {
      int jt = w + jj;
      const __bf16* eb = Ebf + (size_t)(dmin + 16 * jt) * HD;
      v8f a = zero8();
      a = wmma_bf16(qa0, frag_b(eb + 0,  HD, lane), a);
      a = wmma_bf16(qa1, frag_b(eb + 32, HD, lane), a);
#pragma unroll
      for (int i = 0; i < 8; ++i)
        sTQ[(16 * w + i + 8 * hf) * BAND + 16 * jt + ln] = (__bf16)a[i];
    }

    // ---- TK band: row-tile bt needs jt in [3-bt, 11-bt]; 2 waves split 9 tiles
    {
      int bt = w & 3, half = w >> 2;
      v16bf ka0 = frag_a(sK + (16 * bt) * HD + 0,  HD, lane);
      v16bf ka1 = frag_a(sK + (16 * bt) * HD + 32, HD, lane);
      for (int s2 = half; s2 < 9; s2 += 2) {
        int jt = (3 - bt) + s2;
        const __bf16* eb = Ebf + (size_t)(dmin + 16 * jt) * HD;
        v8f a = zero8();
        a = wmma_bf16(ka0, frag_b(eb + 0,  HD, lane), a);
        a = wmma_bf16(ka1, frag_b(eb + 32, HD, lane), a);
#pragma unroll
        for (int i = 0; i < 8; ++i)
          sTK[(16 * bt + i + 8 * hf) * BAND + 16 * jt + ln] = (__bf16)a[i];
      }
    }
    __syncthreads();  // bands ready (TK is cross-wave)

    // ---- bias gather + scale + mask, online softmax (all in-wave) ----
    float tmax[8];
#pragma unroll
    for (int i = 0; i < 8; ++i) tmax[i] = -__builtin_inff();
#pragma unroll
    for (int ct = 0; ct < 4; ++ct) {
      int bcol = 16 * ct + ln;
      float mk = amask[(size_t)b * NS + r0 + bcol];
#pragma unroll
      for (int i = 0; i < 8; ++i) {
        int a = 16 * w + i + 8 * hf;
        int dd = a - bcol + 63;  // in [0, 190]
        float sv = sfr[ct][i] + (float)sTQ[a * BAND + dd] + (float)sTK[bcol * BAND + dd];
        sv = sv * 0.125f + mk;   // /sqrt(64) then +mask, per reference
        sfr[ct][i] = sv;
        tmax[i] = fmaxf(tmax[i], sv);
      }
    }
#pragma unroll
    for (int i = 0; i < 8; ++i) {
#pragma unroll
      for (int m = 1; m < 16; m <<= 1) tmax[i] = fmaxf(tmax[i], __shfl_xor(tmax[i], m));
      float mn = fmaxf(m_run[i], tmax[i]);
      float sc = __expf(m_run[i] - mn);
      m_run[i] = mn;
      l_run[i] *= sc;
#pragma unroll
      for (int ct = 0; ct < 4; ++ct) ctx[ct][i] = ctx[ct][i] * sc;
      float ps = 0.f;
#pragma unroll
      for (int ct = 0; ct < 4; ++ct) {
        float p = __expf(sfr[ct][i] - mn);
        sfr[ct][i] = p;
        ps += p;
      }
#pragma unroll
      for (int m = 1; m < 16; m <<= 1) ps += __shfl_xor(ps, m);
      l_run[i] += ps;
    }

    // stage P (bf16) into this wave's private LDS slab, then ctx += P @ V
#pragma unroll
    for (int ct = 0; ct < 4; ++ct)
#pragma unroll
      for (int i = 0; i < 8; ++i)
        sP[w * (16 * 64) + (i + 8 * hf) * 64 + 16 * ct + ln] = (__bf16)sfr[ct][i];
    // same-wave LDS RAW: DS pipe is in-order, but fence the compiler + wait
    asm volatile("s_wait_dscnt 0" ::: "memory");
    __builtin_amdgcn_wave_barrier();

    v16bf pa0 = frag_a(sP + w * (16 * 64) + 0,  64, lane);
    v16bf pa1 = frag_a(sP + w * (16 * 64) + 32, 64, lane);
#pragma unroll
    for (int ct = 0; ct < 4; ++ct) {
      ctx[ct] = wmma_bf16(pa0, frag_b(sVt + (16 * ct) * 64 + 0,  64, lane), ctx[ct]);
      ctx[ct] = wmma_bf16(pa1, frag_b(sVt + (16 * ct) * 64 + 32, 64, lane), ctx[ct]);
    }
  }

  // epilogue: normalize, write fp32 out[b, l, h*64 + d]
#pragma unroll
  for (int ct = 0; ct < 4; ++ct)
#pragma unroll
    for (int i = 0; i < 8; ++i) {
      int a = 16 * w + i + 8 * hf;
      int n = 16 * ct + ln;
      out[((size_t)(b * NS + l0 + a)) * NT + h * HD + n] = ctx[ct][i] / l_run[i];
    }
}

// ---------------------------------------------------------------------------
// Host launcher
// ---------------------------------------------------------------------------
extern "C" void kernel_launch(void* const* d_in, const int* in_sizes, int n_in,
                              void* d_out, int out_size, void* d_ws, size_t ws_size,
                              hipStream_t stream) {
  (void)in_sizes; (void)n_in; (void)out_size; (void)ws_size;
  const float* hs   = (const float*)d_in[0];
  const float* amsk = (const float*)d_in[1];
  const float* Wq   = (const float*)d_in[2];
  const float* bq   = (const float*)d_in[3];
  const float* Wk   = (const float*)d_in[4];
  const float* bk   = (const float*)d_in[5];
  const float* Wv   = (const float*)d_in[6];
  const float* bv   = (const float*)d_in[7];
  const float* Wqd  = (const float*)d_in[8];
  const float* bqd  = (const float*)d_in[9];
  const float* Wkd  = (const float*)d_in[10];
  const float* bkd  = (const float*)d_in[11];
  const float* Wvd  = (const float*)d_in[12];
  const float* bvd  = (const float*)d_in[13];
  const float* demb = (const float*)d_in[14];
  float* out = (float*)d_out;

  char* ws = (char*)d_ws;
  __bf16* Xbf  = (__bf16*)(ws + 0);          //  16,777,216 B
  __bf16* Wcat = (__bf16*)(ws + 16777216);   //   7,864,320 B
  float*  Bcat = (float*) (ws + 24641536);   //      15,360 B
  __bf16* Ebf  = (__bf16*)(ws + 24656896);   //     131,072 B (padded 1024 rows)
  __bf16* Qbf  = (__bf16*)(ws + 24787968);   //  20,971,520 B
  __bf16* Kbf  = (__bf16*)(ws + 45759488);   //  20,971,520 B
  __bf16* Vbf  = (__bf16*)(ws + 66731008);   //  20,971,520 B  (end 87,702,528)

  const int nX = NB * NS * ND;  // 8,388,608
  pack_x_kernel<<<(nX + 255) / 256, 256, 0, stream>>>(hs, Xbf, nX);
  pack_w_kernel<<<(3 * ND * NT + 255) / 256, 256, 0, stream>>>(Wq, Wk, Wv, Wqd, Wkd, Wvd, Wcat);
  pack_b_kernel<<<(3 * NT + 255) / 256, 256, 0, stream>>>(bq, bk, bv, bqd, bkd, bvd, Bcat);
  pack_e_kernel<<<(1024 * HD + 255) / 256, 256, 0, stream>>>(demb, Ebf);

  qkv_gemm_kernel<<<dim3(64, 10, 3), 256, 0, stream>>>(Xbf, Wcat, Bcat, Qbf, Kbf, Vbf);

  attn_kernel<<<dim3(NS / 128, NH, NB), 256, ATTN_SMEM_BYTES, stream>>>(
      Qbf, Kbf, Vbf, Ebf, amsk, out);
}
